// main_net_86431921865190
// MI455X (gfx1250) — matrix operand order repaired and checked by reading, not compiled, not run on hardware
//
#include <hip/hip_runtime.h>

// Problem constants (match reference)
#define NPIXC 3072      // 12 * 16 * 16
#define BATCH 16
#define HIDC  64
#define OUTD  100
#define NELEM 2000000
#define LN_EPSF 1e-5f
#define GN (BATCH * HIDC)   // 1024 : columns of the big GEMM

typedef __attribute__((ext_vector_type(16))) _Float16 v16h;
typedef __attribute__((ext_vector_type(8)))  _Float16 v8h;
typedef __attribute__((ext_vector_type(8)))  float    v8f;

// ---------------------------------------------------------------------------
// Kernel 1: scatter-mean partials. LDS-privatized histogram per block.
// grid = (125, BATCH), block = 256.  125*16000 == NELEM exactly.
// ---------------------------------------------------------------------------
__global__ __launch_bounds__(256) void scatter_kernel(
    const float* __restrict__ x, const int* __restrict__ pix,
    float* __restrict__ sumsT, float* __restrict__ cntsT) {
  __shared__ float s[NPIXC];
  __shared__ float c[NPIXC];
  const int b = blockIdx.y;
  const int per = NELEM / 125;              // 16000
  const int start = blockIdx.x * per;
  const int end = start + per;
  for (int j = threadIdx.x; j < NPIXC; j += 256) { s[j] = 0.f; c[j] = 0.f; }
  __syncthreads();
  const float* vals = x + (size_t)b * 3 * NELEM + 2 * (size_t)NELEM;
  const int* pp = pix + (size_t)b * NELEM;
  for (int i = start + threadIdx.x; i < end; i += 256) {
    int p = pp[i];
    atomicAdd(&s[p], vals[i]);
    atomicAdd(&c[p], 1.0f);
  }
  __syncthreads();
  for (int p = threadIdx.x; p < NPIXC; p += 256) {
    atomicAdd(&sumsT[p * BATCH + b], s[p]);
    atomicAdd(&cntsT[p * BATCH + b], c[p]);
  }
}

// ---------------------------------------------------------------------------
// Kernel 2: pooled = sum / max(count,1), stored transposed [p][b] in place.
// ---------------------------------------------------------------------------
__global__ void finalize_kernel(float* __restrict__ sumsT,
                                const float* __restrict__ cntsT) {
  int i = blockIdx.x * blockDim.x + threadIdx.x;
  if (i < NPIXC * BATCH) {
    float cnt = cntsT[i];
    if (cnt == 0.f) cnt = 1.f;
    sumsT[i] = sumsT[i] / cnt;
  }
}

// ---------------------------------------------------------------------------
// Kernel 3: scalar stats of W1/b1 for the analytic conv1 layernorm.
// ---------------------------------------------------------------------------
__global__ void stats_kernel(const float* __restrict__ W1,
                             const float* __restrict__ b1,
                             float* __restrict__ stats) {
  float mw = 0.f, mc = 0.f;
  for (int f = 0; f < HIDC; ++f) { mw += W1[f]; mc += b1[f]; }
  mw *= (1.f / HIDC); mc *= (1.f / HIDC);
  float sww = 0.f, swc = 0.f, scc = 0.f;
  for (int f = 0; f < HIDC; ++f) {
    float w = W1[f] - mw, cc = b1[f] - mc;
    sww += w * w; swc += w * cc; scc += cc * cc;
  }
  stats[0] = mw; stats[1] = mc;
  stats[2] = sww * (1.f / HIDC);
  stats[3] = swc * (1.f / HIDC);
  stats[4] = scc * (1.f / HIDC);
}

// ---------------------------------------------------------------------------
// Kernel 4: conv1. One block per adj row p.
//  - agg[b] = sum_q adj[p][q] * poolT[q][b]  (exploits ~8 nnz/row)
//  - also converts adj row to f16 (GEMM A matrix)
//  - analytic LN + relu -> h1 written as Bq[q=p][b*64+f] in f16 (GEMM B matrix)
// ---------------------------------------------------------------------------
__global__ __launch_bounds__(256) void conv1_kernel(
    const float* __restrict__ adj, const float* __restrict__ poolT,
    const float* __restrict__ W1, const float* __restrict__ b1,
    const float* __restrict__ g1, const float* __restrict__ be1,
    const float* __restrict__ stats,
    _Float16* __restrict__ adjH, _Float16* __restrict__ Bq) {
  __shared__ float red[BATCH][256];
  const int p = blockIdx.x;
  const float* row = adj + (size_t)p * NPIXC;
  float acc[BATCH];
#pragma unroll
  for (int b = 0; b < BATCH; ++b) acc[b] = 0.f;
  for (int q = threadIdx.x; q < NPIXC; q += 256) {
    float av = row[q];
    adjH[(size_t)p * NPIXC + q] = (_Float16)av;
    if (av != 0.0f) {
      const float* pt = poolT + q * BATCH;
#pragma unroll
      for (int b = 0; b < BATCH; ++b) acc[b] += av * pt[b];
    }
  }
#pragma unroll
  for (int b = 0; b < BATCH; ++b) red[b][threadIdx.x] = acc[b];
  __syncthreads();
  for (int off = 128; off > 0; off >>= 1) {
    if (threadIdx.x < off) {
#pragma unroll
      for (int b = 0; b < BATCH; ++b)
        red[b][threadIdx.x] += red[b][threadIdx.x + off];
    }
    __syncthreads();
  }
  const float mw = stats[0], mc = stats[1];
  const float sww = stats[2], swc = stats[3], scc = stats[4];
  for (int idx = threadIdx.x; idx < GN; idx += 256) {
    int b = idx >> 6, f = idx & 63;
    float a = red[b][0];
    float o = a * W1[f] + b1[f];
    float mu = a * mw + mc;
    float var = a * a * sww + 2.f * a * swc + scc;
    float y = (o - mu) * rsqrtf(var + LN_EPSF) * g1[f] + be1[f];
    y = fmaxf(y, 0.f);
    Bq[(size_t)p * GN + idx] = (_Float16)y;
  }
}

// ---------------------------------------------------------------------------
// Kernel 5: WMMA GEMM: agg2(3072 x 1024) = adjH(3072 x 3072) * Bq(3072 x 1024)
// f16 inputs, f32 accumulate via v_wmma_f32_16x16x32_f16.
// Block tile 128x64, 8 waves (wave w -> rows [16w,16w+16), 4 N-tiles), K-step 32.
// grid = (3072/128, 1024/64) = (24, 16), block = 256.
// ---------------------------------------------------------------------------
__global__ __launch_bounds__(256) void gemm_kernel(
    const _Float16* __restrict__ A, const _Float16* __restrict__ B,
    float* __restrict__ C) {
  __shared__ __align__(16) _Float16 Alds[128 * 40];  // 128 rows x 32 halfs, pad->40
  __shared__ __align__(16) _Float16 Blds[64 * 40];   // B^T: 64 cols x 32 halfs, pad->40
  const int t = threadIdx.x;
  const int wave = t >> 5;
  const int lane = t & 31;
  const int pBase = blockIdx.x * 128;
  const int nBase = blockIdx.y * 64;
  v8f cacc[4] = {};
  const int lrow = lane & 15;
  const int kbA = (lane >> 4) * 8;    // A: lane-halves alternate 8-K blocks
  const int ksB = (lane >> 4) * 16;   // unused: B fragment also uses kbA (8-K blocks)

  for (int k0 = 0; k0 < NPIXC; k0 += 32) {
    __syncthreads();
    // Stage A tile 128x32 (two 16B chunks per thread)
#pragma unroll
    for (int i = 0; i < 2; ++i) {
      int chunk = t + i * 256;           // 512 chunks of 8 halfs
      int r = chunk >> 2;
      int cpos = (chunk & 3) * 8;
      v8h v = *reinterpret_cast<const v8h*>(
          A + (size_t)(pBase + r) * NPIXC + k0 + cpos);
      *reinterpret_cast<v8h*>(&Alds[r * 40 + cpos]) = v;
    }
    // Stage B tile 32x64, transposed into Blds[n][k]
    {
      int k = t >> 3;
      int npos = (t & 7) * 8;
      v8h v = *reinterpret_cast<const v8h*>(
          B + (size_t)(k0 + k) * GN + nBase + npos);
#pragma unroll
      for (int j = 0; j < 8; ++j) Blds[(npos + j) * 40 + k] = v[j];
    }
    __syncthreads();

    // A fragment: row = wave*16 + (lane&15); K = {kbA..kbA+7} U {kbA+16..kbA+23}
    const int arow = wave * 16 + lrow;
    v8h alo = *reinterpret_cast<const v8h*>(&Alds[arow * 40 + kbA]);
    v8h ahi = *reinterpret_cast<const v8h*>(&Alds[arow * 40 + kbA + 16]);
    v16h afrag = __builtin_shufflevector(alo, ahi,
        0, 1, 2, 3, 4, 5, 6, 7, 8, 9, 10, 11, 12, 13, 14, 15);
#pragma unroll
    for (int n = 0; n < 4; ++n) {
      // B fragment: col = n*16 + (lane&15); K = {kbA..kbA+7} U {kbA+16..kbA+23}
      const int bcol = n * 16 + lrow;
      v8h blo = *reinterpret_cast<const v8h*>(&Blds[bcol * 40 + kbA]);
      v8h bhi = *reinterpret_cast<const v8h*>(&Blds[bcol * 40 + kbA + 16]);
      v16h bfrag = __builtin_shufflevector(blo, bhi,
          0, 1, 2, 3, 4, 5, 6, 7, 8, 9, 10, 11, 12, 13, 14, 15);
      cacc[n] = __builtin_amdgcn_wmma_f32_16x16x32_f16(
          false, afrag, false, bfrag, (short)0, cacc[n], false, false);
    }
  }

  // Store: C/D layout -> element (M = r + 8*(lane>>4), N = lane&15) in vgpr r
  const int mhi = (lane >> 4) * 8;
#pragma unroll
  for (int n = 0; n < 4; ++n) {
#pragma unroll
    for (int r = 0; r < 8; ++r) {
      int row = pBase + wave * 16 + mhi + r;
      int col = nBase + n * 16 + lrow;
      C[(size_t)row * GN + col] = cacc[n][r];
    }
  }
}

// ---------------------------------------------------------------------------
// Kernel 6: conv2 epilogue: o = agg2_row @ W2 + b2, LN over 64 feats, relu,
// partial pixel-mean accumulation. grid = (48, BATCH), block = 256 (4 groups
// of 64 threads, each group walks 16 pixels).
// ---------------------------------------------------------------------------
__global__ __launch_bounds__(256) void conv2_kernel(
    const float* __restrict__ agg2, const float* __restrict__ W2,
    const float* __restrict__ b2, const float* __restrict__ g2,
    const float* __restrict__ be2, float* __restrict__ meanAcc) {
  __shared__ float W2s[HIDC * HIDC];
  __shared__ float rowbuf[4][HIDC];
  __shared__ float redA[4][HIDC];
  __shared__ float redB[4][HIDC];
  const int t = threadIdx.x;
  for (int i = t; i < HIDC * HIDC; i += 256) W2s[i] = W2[i];
  const int g = t >> 6;
  const int f = t & 63;
  const int b = blockIdx.y;
  const int pBase = blockIdx.x * 64 + g * 16;
  const float bb = b2[f], gg = g2[f], bee = be2[f];
  float accum = 0.f;
  for (int it = 0; it < 16; ++it) {
    const int p = pBase + it;
    __syncthreads();  // covers W2s staging (it==0) and prior rowbuf readers
    rowbuf[g][f] = agg2[(size_t)p * GN + b * HIDC + f];
    __syncthreads();
    float o = bb;
#pragma unroll
    for (int fp = 0; fp < HIDC; ++fp) o += rowbuf[g][fp] * W2s[fp * HIDC + f];
    redA[g][f] = o;
    redB[g][f] = o * o;
    __syncthreads();
    for (int off = 32; off > 0; off >>= 1) {
      if (f < off) {
        redA[g][f] += redA[g][f + off];
        redB[g][f] += redB[g][f + off];
      }
      __syncthreads();
    }
    float mu = redA[g][0] * (1.f / HIDC);
    float var = redB[g][0] * (1.f / HIDC) - mu * mu;
    float y = (o - mu) * rsqrtf(fmaxf(var, 0.f) + LN_EPSF) * gg + bee;
    accum += fmaxf(y, 0.f);
  }
  atomicAdd(&meanAcc[b * HIDC + f], accum * (1.0f / NPIXC));
}

// ---------------------------------------------------------------------------
// Kernel 7: readout: out[b][j] = (meanAcc[b] . Wr[:,j] + br[j]) * scale
// ---------------------------------------------------------------------------
__global__ void readout_kernel(const float* __restrict__ meanAcc,
                               const float* __restrict__ Wr,
                               const float* __restrict__ br,
                               const float* __restrict__ scale,
                               float* __restrict__ out) {
  int idx = blockIdx.x * blockDim.x + threadIdx.x;
  if (idx >= BATCH * OUTD) return;
  int b = idx / OUTD, j = idx % OUTD;
  float acc = br[j];
  const float* m = meanAcc + b * HIDC;
  for (int fIt = 0; fIt < HIDC; ++fIt) acc += m[fIt] * Wr[fIt * OUTD + j];
  out[idx] = acc * scale[0];
}

// ---------------------------------------------------------------------------
extern "C" void kernel_launch(void* const* d_in, const int* in_sizes, int n_in,
                              void* d_out, int out_size, void* d_ws,
                              size_t ws_size, hipStream_t stream) {
  (void)in_sizes; (void)n_in; (void)out_size; (void)ws_size;
  const float* x     = (const float*)d_in[0];
  const int*   pix   = (const int*)d_in[1];
  const float* adj   = (const float*)d_in[2];
  const float* W1    = (const float*)d_in[3];
  const float* b1    = (const float*)d_in[4];
  const float* g1    = (const float*)d_in[5];
  const float* be1   = (const float*)d_in[6];
  const float* W2    = (const float*)d_in[7];
  const float* b2    = (const float*)d_in[8];
  const float* g2    = (const float*)d_in[9];
  const float* be2   = (const float*)d_in[10];
  const float* Wr    = (const float*)d_in[11];
  const float* br    = (const float*)d_in[12];
  const float* scale = (const float*)d_in[13];
  float* out = (float*)d_out;

  char* ws = (char*)d_ws;
  size_t off = 0;
  auto alloc = [&](size_t bytes) -> void* {
    void* p = ws + off;
    off = (off + bytes + 255) & ~(size_t)255;
    return p;
  };
  float*     sumsT   = (float*)alloc((size_t)NPIXC * BATCH * 4);  // -> poolT
  float*     cntsT   = (float*)alloc((size_t)NPIXC * BATCH * 4);
  float*     statsB  = (float*)alloc(8 * 4);
  float*     meanAcc = (float*)alloc((size_t)BATCH * HIDC * 4);
  _Float16*  adjH    = (_Float16*)alloc((size_t)NPIXC * NPIXC * 2);
  _Float16*  Bq      = (_Float16*)alloc((size_t)NPIXC * GN * 2);
  float*     agg2    = (float*)alloc((size_t)NPIXC * GN * 4);

  hipMemsetAsync(sumsT, 0, (size_t)NPIXC * BATCH * 4, stream);
  hipMemsetAsync(cntsT, 0, (size_t)NPIXC * BATCH * 4, stream);
  hipMemsetAsync(meanAcc, 0, (size_t)BATCH * HIDC * 4, stream);

  scatter_kernel<<<dim3(125, BATCH), 256, 0, stream>>>(x, pix, sumsT, cntsT);
  finalize_kernel<<<(NPIXC * BATCH + 255) / 256, 256, 0, stream>>>(sumsT, cntsT);
  stats_kernel<<<1, 1, 0, stream>>>(W1, b1, statsB);
  conv1_kernel<<<NPIXC, 256, 0, stream>>>(adj, sumsT, W1, b1, g1, be1, statsB,
                                          adjH, Bq);
  gemm_kernel<<<dim3(NPIXC / 128, GN / 64), 256, 0, stream>>>(adjH, Bq, agg2);
  conv2_kernel<<<dim3(NPIXC / 64, BATCH), 256, 0, stream>>>(agg2, W2, b2, g2,
                                                            be2, meanAcc);
  readout_kernel<<<(BATCH * OUTD + 255) / 256, 256, 0, stream>>>(
      meanAcc, Wr, br, scale, out);
}
